// BidRNN_3977139716151
// MI455X (gfx1250) — matrix operand order repaired and checked by reading, not compiled, not run on hardware
//
#include <hip/hip_runtime.h>
#include <hip/hip_bf16.h>

// ---------------------------------------------------------------------------
// BidRNN for MI455X (gfx1250): bf16 WMMA + async global->LDS staging.
//   VOCAB=32000, EMBED=256, HIDDEN=512, B=16, T=512
// ---------------------------------------------------------------------------

#define VOCAB  32000
#define EMBED  256
#define HIDDEN 512
#define BATCH  16
#define TLEN   512
#define HID2   (2*HIDDEN)

typedef __attribute__((ext_vector_type(16))) __bf16        v16bf;
typedef __attribute__((ext_vector_type(8)))  float         v8f;
typedef __attribute__((ext_vector_type(8)))  unsigned int  v8u;
typedef __attribute__((ext_vector_type(4)))  int           v4i;

#if __has_builtin(__builtin_amdgcn_global_load_async_to_lds_b128) && \
    __has_builtin(__builtin_amdgcn_s_wait_asynccnt)
#define HAS_ASYNC 1
#else
#define HAS_ASYNC 0
#endif

#if HAS_ASYNC
typedef __attribute__((address_space(1))) v4i g_v4i;   // global int4
typedef __attribute__((address_space(3))) v4i l_v4i;   // LDS int4
// flat LDS addresses keep the byte offset in the low 32 bits (ISA 10.2)
#define TO_GLB(p) ((g_v4i*)(size_t)(p))
#define TO_LDS(p) ((l_v4i*)(unsigned int)(size_t)(p))
#define ASYNC_CP16(g, l) \
    __builtin_amdgcn_global_load_async_to_lds_b128(TO_GLB(g), TO_LDS(l), 0, 0)
#endif

// float -> bf16 bits, round-to-nearest-even
__device__ __forceinline__ unsigned short f2bf(float f) {
    unsigned int u = __builtin_bit_cast(unsigned int, f);
    unsigned int r = u + 0x7FFFu + ((u >> 16) & 1u);
    return (unsigned short)(r >> 16);
}

// ---- WMMA bf16 fragment loaders (ISA 7.12.2 layouts) ----------------------
// A matrix 16x32 (MxK), row-major bf16 source, leading dim `ld` elements.
__device__ __forceinline__ v16bf load_a_frag(const unsigned short* src,
                                             int row0, int ld, int k0) {
    int lane = threadIdx.x & 31;
    int half = lane >> 4;
    int m    = lane & 15;
    const unsigned short* rowp = src + (size_t)(row0 + m) * ld + k0 + half * 8;
    v8u u;
#pragma unroll
    for (int v = 0; v < 4; ++v)
        u[v] = *(const unsigned int*)(rowp + 2 * v);
#pragma unroll
    for (int v = 0; v < 4; ++v)
        u[4 + v] = *(const unsigned int*)(rowp + 16 + 2 * v);
    return __builtin_bit_cast(v16bf, u);
}

// B matrix 32x16 (KxN) fragment from PRE-TRANSPOSED weights (N x K row-major).
__device__ __forceinline__ v16bf load_b_frag(const unsigned short* wt,
                                             int col0, int ld, int k0) {
    int lane = threadIdx.x & 31;
    int half = lane >> 4;
    int n    = lane & 15;
    const unsigned short* rowp = wt + (size_t)(col0 + n) * ld + k0 + half * 8;
    v8u u;
#pragma unroll
    for (int v = 0; v < 8; ++v)
        u[v] = *(const unsigned int*)(rowp + 2 * v + (v >> 2) * 8);
    return __builtin_bit_cast(v16bf, u);
}

#define WMMA_BF16(a, b, c) \
    __builtin_amdgcn_wmma_f32_16x16x32_bf16(false, (a), false, (b), (short)0, (c), false, false)

// ---------------------------------------------------------------------------
// K1: embedding gather + f32->bf16. x stored (T*B, E) row-major, row = t*B+b.
// ---------------------------------------------------------------------------
__global__ void k_embed(const int* __restrict__ tok,
                        const float* __restrict__ emb,
                        unsigned short* __restrict__ xb) {
    int idx = blockIdx.x * blockDim.x + threadIdx.x;
    if (idx >= TLEN * BATCH * EMBED) return;
    int e = idx & (EMBED - 1);
    int r = idx >> 8;
    int t = r >> 4;
    int b = r & 15;
    int token = tok[b * TLEN + t];
    xb[idx] = f2bf(emb[(size_t)token * EMBED + e]);
}

// ---------------------------------------------------------------------------
// K2: transpose + convert f32 (K x N) -> bf16 (N x K)
// ---------------------------------------------------------------------------
__global__ void k_transpose_bf16(const float* __restrict__ in,
                                 unsigned short* __restrict__ out,
                                 int K, int N, long total) {
    long idx = (long)blockIdx.x * blockDim.x + threadIdx.x;
    if (idx >= total) return;
    long k = idx / N;
    long n = idx - k * N;
    out[n * K + k] = f2bf(in[k * N + n]);
}

// ---------------------------------------------------------------------------
// K3: xW = x @ W_xh + b_h   (8192 x 256) @ (256 x 512), f32 out (T,B,H).
// ---------------------------------------------------------------------------
__global__ void k_gemm_xw(const unsigned short* __restrict__ xb,
                          const unsigned short* __restrict__ wxhT,
                          const float* __restrict__ bias,
                          float* __restrict__ xw) {
    int wave = threadIdx.x >> 5;
    int row0 = blockIdx.x * 16;
    int col0 = (blockIdx.y * 4 + wave) * 16;
    int lane = threadIdx.x & 31;
    int n    = lane & 15;
    v8f acc = {};
#pragma unroll
    for (int k0 = 0; k0 < EMBED; k0 += 32) {
        __builtin_prefetch(wxhT + (size_t)(col0 + n) * EMBED + k0 + 64, 0, 3);
        v16bf a = load_a_frag(xb, row0, EMBED, k0);
        v16bf b = load_b_frag(wxhT, col0, EMBED, k0);
        acc = WMMA_BF16(a, b, acc);
    }
    int half = lane >> 4;
    float bv = bias[col0 + n];
#pragma unroll
    for (int i = 0; i < 8; ++i) {
        int m = half * 8 + i;
        xw[(size_t)(row0 + m) * HIDDEN + col0 + n] = acc[i] + bv;
    }
}

// ---------------------------------------------------------------------------
// K4: sequential recurrence. One workgroup per direction, 1024 threads =
// 32 waves; wave w owns output columns [16w,16w+16). h in LDS (16x512 bf16).
// Next step's xW tile (32KB) is async-prefetched into LDS during the WMMAs.
// ---------------------------------------------------------------------------
__global__ void k_rnn(const unsigned short* __restrict__ whhT_f,
                      const unsigned short* __restrict__ whhT_b,
                      const float* __restrict__ xw_f,
                      const float* __restrict__ xw_b,
                      unsigned short* __restrict__ hcat) {
    __shared__ unsigned short hl[BATCH * HIDDEN];          // 16 KB
#if HAS_ASYNC
    __shared__ float xls[2][BATCH * HIDDEN];               // 64 KB dbl-buffer
#endif
    const int dir  = blockIdx.x;
    const unsigned short* whhT = dir ? whhT_b : whhT_f;
    const float*          xw   = dir ? xw_b   : xw_f;

    const int tid  = threadIdx.x;
    const int wave = tid >> 5;
    const int lane = tid & 31;
    const int half = lane >> 4;
    const int mn   = lane & 15;
    const int col0 = wave * 16;

    for (int i = tid; i < BATCH * HIDDEN; i += 1024) hl[i] = 0;   // h_prev = 0

#if HAS_ASYNC
    // prologue: stage xW tile for the first step (2048 x 16B chunks, 2/thread)
    {
        int t0 = dir ? (TLEN - 1) : 0;
        const float* g = xw + (size_t)t0 * BATCH * HIDDEN;
#pragma unroll
        for (int j = 0; j < 2; ++j) {
            int c = tid + j * 1024;
            ASYNC_CP16(g + c * 4, &xls[0][c * 4]);
        }
    }
#endif
    __syncthreads();

    for (int s = 0; s < TLEN; ++s) {
        int t = dir ? (TLEN - 1 - s) : s;
        int cur = s & 1;
#if HAS_ASYNC
        if (s + 1 < TLEN) {            // stage next step's xW while we compute
            int tn = dir ? (TLEN - 2 - s) : (s + 1);
            const float* g = xw + (size_t)tn * BATCH * HIDDEN;
#pragma unroll
            for (int j = 0; j < 2; ++j) {
                int c = tid + j * 1024;
                ASYNC_CP16(g + c * 4, &xls[cur ^ 1][c * 4]);
            }
        }
#endif
        v8f acc = {};
#pragma unroll 4
        for (int ks = 0; ks < HIDDEN / 32; ++ks) {
            v16bf a = load_a_frag(hl, 0, HIDDEN, ks * 32);      // ds path
            v16bf b = load_b_frag(whhT, col0, HIDDEN, ks * 32); // L2 stream
            acc = WMMA_BF16(a, b, acc);
        }
#if HAS_ASYNC
        if (s + 1 < TLEN) __builtin_amdgcn_s_wait_asynccnt(2);  // this step done
        else              __builtin_amdgcn_s_wait_asynccnt(0);
#endif
        __syncthreads();   // all waves: done reading old h AND xls[cur] resident

        float hv[8];
#pragma unroll
        for (int i = 0; i < 8; ++i) {
            int m = half * 8 + i;                               // batch index
#if HAS_ASYNC
            float pre = acc[i] + xls[cur][m * HIDDEN + col0 + mn];
#else
            float pre = acc[i] + xw[(size_t)(t * BATCH + m) * HIDDEN + col0 + mn];
#endif
            hv[i] = tanhf(pre);
        }
#pragma unroll
        for (int i = 0; i < 8; ++i) {
            int m = half * 8 + i;
            unsigned short hb = f2bf(hv[i]);
            hl[m * HIDDEN + col0 + mn] = hb;
            hcat[((size_t)m * TLEN + t) * HID2 + dir * HIDDEN + col0 + mn] = hb;
        }
        __syncthreads();   // new h visible to all waves
    }
}

// ---------------------------------------------------------------------------
// K5: out = h_cat @ W_fc + b_fc   (8192 x 1024) @ (1024 x 32000), f32 out.
// Block 256 threads (8 waves), block tile 128M x 128N, KC=32, double-buffered
// async global->LDS staging of both A and B. Wave tile 32M x 64N (8 WMMAs per
// k-chunk from 6 LDS fragment loads).
// ---------------------------------------------------------------------------
__global__ void k_gemm_fc(const unsigned short* __restrict__ hc,
                          const unsigned short* __restrict__ wfcT,
                          const float* __restrict__ bfc,
                          float* __restrict__ out) {
    __shared__ unsigned short As[2][128 * 32];   // 8 KB each
    __shared__ unsigned short Bs[2][128 * 32];   // 8 KB each

    const int tid    = threadIdx.x;
    const int wave   = tid >> 5;
    const int wave_m = wave & 3;                 // 0..3 -> 32-row strip
    const int wave_n = wave >> 2;                // 0..1 -> 64-col strip
    const int lane   = tid & 31;
    const int half   = lane >> 4;
    const int n      = lane & 15;
    const int row0   = blockIdx.y * 128;
    const int col0   = blockIdx.x * 128;

    v8f acc[2][4];
#pragma unroll
    for (int mi = 0; mi < 2; ++mi)
#pragma unroll
        for (int ni = 0; ni < 4; ++ni) acc[mi][ni] = {};

#if HAS_ASYNC
    // async copy of one 128x32 bf16 tile: 512 x 16B chunks, 2 per thread
#define CP_TILE(gbase, grow0, buf, smem, k0)                                   \
    {                                                                          \
        _Pragma("unroll")                                                      \
        for (int j = 0; j < 2; ++j) {                                          \
            int c = tid + j * 256;                                             \
            int rr = c >> 2, sub = c & 3;                                      \
            ASYNC_CP16(gbase + (size_t)(grow0 + rr) * HID2 + (k0) + sub * 8,   \
                       &smem[buf][rr * 32 + sub * 8]);                         \
        }                                                                      \
    }
    CP_TILE(hc,   row0, 0, As, 0)
    CP_TILE(wfcT, col0, 0, Bs, 0)
    CP_TILE(hc,   row0, 1, As, 32)
    CP_TILE(wfcT, col0, 1, Bs, 32)

#pragma unroll 2
    for (int i = 0; i < HID2 / 32; ++i) {
        int cur = i & 1;
        if (i < HID2 / 32 - 1) __builtin_amdgcn_s_wait_asynccnt(8); // chunk i resident
        else                   __builtin_amdgcn_s_wait_asynccnt(0);
        __syncthreads();
        const unsigned short* Ab = &As[cur][0];
        const unsigned short* Bb = &Bs[cur][0];
        v16bf a0 = load_a_frag(Ab, wave_m * 32, 32, 0);
        v16bf a1 = load_a_frag(Ab, wave_m * 32 + 16, 32, 0);
#pragma unroll
        for (int ni = 0; ni < 4; ++ni) {
            v16bf b = load_b_frag(Bb, wave_n * 64 + ni * 16, 32, 0);
            acc[0][ni] = WMMA_BF16(a0, b, acc[0][ni]);
            acc[1][ni] = WMMA_BF16(a1, b, acc[1][ni]);
        }
        __syncthreads();
        if (i + 2 < HID2 / 32) {
            CP_TILE(hc,   row0, cur, As, (i + 2) * 32)
            CP_TILE(wfcT, col0, cur, Bs, (i + 2) * 32)
        }
    }
#undef CP_TILE
#else
    // fallback: synchronous staging through VGPRs, single buffer
    for (int i = 0; i < HID2 / 32; ++i) {
        __syncthreads();
#pragma unroll
        for (int j = 0; j < 2; ++j) {
            int c = tid + j * 256;
            int rr = c >> 2, sub = c & 3;
            *(uint4*)&As[0][rr * 32 + sub * 8] =
                *(const uint4*)(hc + (size_t)(row0 + rr) * HID2 + i * 32 + sub * 8);
            *(uint4*)&Bs[0][rr * 32 + sub * 8] =
                *(const uint4*)(wfcT + (size_t)(col0 + rr) * HID2 + i * 32 + sub * 8);
        }
        __syncthreads();
        v16bf a0 = load_a_frag(&As[0][0], wave_m * 32, 32, 0);
        v16bf a1 = load_a_frag(&As[0][0], wave_m * 32 + 16, 32, 0);
#pragma unroll
        for (int ni = 0; ni < 4; ++ni) {
            v16bf b = load_b_frag(&Bs[0][0], wave_n * 64 + ni * 16, 32, 0);
            acc[0][ni] = WMMA_BF16(a0, b, acc[0][ni]);
            acc[1][ni] = WMMA_BF16(a1, b, acc[1][ni]);
        }
    }
#endif

#pragma unroll
    for (int mi = 0; mi < 2; ++mi) {
#pragma unroll
        for (int i = 0; i < 8; ++i) {
            int m = wave_m * 32 + mi * 16 + half * 8 + i;
            size_t base = (size_t)(row0 + m) * VOCAB + col0 + wave_n * 64;
#pragma unroll
            for (int ni = 0; ni < 4; ++ni)
                out[base + ni * 16 + n] = acc[mi][ni][i] + bfc[col0 + wave_n * 64 + ni * 16 + n];
        }
    }
}

// ---------------------------------------------------------------------------
// Host: workspace layout + launches (all on `stream`, graph-capture safe).
// ---------------------------------------------------------------------------
extern "C" void kernel_launch(void* const* d_in, const int* in_sizes, int n_in,
                              void* d_out, int out_size, void* d_ws, size_t ws_size,
                              hipStream_t stream) {
    const int*   tok   = (const int*)  d_in[0];
    const float* emb   = (const float*)d_in[2];
    const float* Wxh_f = (const float*)d_in[3];
    const float* Whh_f = (const float*)d_in[4];
    const float* bh_f  = (const float*)d_in[5];
    const float* Wxh_b = (const float*)d_in[6];
    const float* Whh_b = (const float*)d_in[7];
    const float* bh_b  = (const float*)d_in[8];
    const float* Wfc   = (const float*)d_in[9];
    const float* bfc   = (const float*)d_in[10];

    char* ws = (char*)d_ws;
    size_t off = 0;
    unsigned short* xb      = (unsigned short*)(ws + off); off += (size_t)TLEN*BATCH*EMBED*2;
    unsigned short* wxhT_f  = (unsigned short*)(ws + off); off += (size_t)HIDDEN*EMBED*2;
    unsigned short* wxhT_b  = (unsigned short*)(ws + off); off += (size_t)HIDDEN*EMBED*2;
    unsigned short* whhT_f  = (unsigned short*)(ws + off); off += (size_t)HIDDEN*HIDDEN*2;
    unsigned short* whhT_b  = (unsigned short*)(ws + off); off += (size_t)HIDDEN*HIDDEN*2;
    unsigned short* wfcT    = (unsigned short*)(ws + off); off += (size_t)VOCAB*HID2*2;
    float*          xw_f    = (float*)(ws + off);          off += (size_t)TLEN*BATCH*HIDDEN*4;
    float*          xw_b    = (float*)(ws + off);          off += (size_t)TLEN*BATCH*HIDDEN*4;
    unsigned short* hcat    = (unsigned short*)(ws + off); off += (size_t)BATCH*TLEN*HID2*2;

    {
        long total = (long)TLEN * BATCH * EMBED;
        k_embed<<<dim3((total + 255) / 256), dim3(256), 0, stream>>>(tok, emb, xb);
    }
    {
        long t1 = (long)EMBED * HIDDEN;
        k_transpose_bf16<<<dim3((t1 + 255) / 256), dim3(256), 0, stream>>>(Wxh_f, wxhT_f, EMBED, HIDDEN, t1);
        k_transpose_bf16<<<dim3((t1 + 255) / 256), dim3(256), 0, stream>>>(Wxh_b, wxhT_b, EMBED, HIDDEN, t1);
        long t2 = (long)HIDDEN * HIDDEN;
        k_transpose_bf16<<<dim3((t2 + 255) / 256), dim3(256), 0, stream>>>(Whh_f, whhT_f, HIDDEN, HIDDEN, t2);
        k_transpose_bf16<<<dim3((t2 + 255) / 256), dim3(256), 0, stream>>>(Whh_b, whhT_b, HIDDEN, HIDDEN, t2);
        long t3 = (long)HID2 * VOCAB;
        k_transpose_bf16<<<dim3((t3 + 255) / 256), dim3(256), 0, stream>>>(Wfc, wfcT, HID2, VOCAB, t3);
    }
    k_gemm_xw<<<dim3(TLEN * BATCH / 16, HIDDEN / 64), dim3(128), 0, stream>>>(xb, wxhT_f, bh_f, xw_f);
    k_gemm_xw<<<dim3(TLEN * BATCH / 16, HIDDEN / 64), dim3(128), 0, stream>>>(xb, wxhT_b, bh_b, xw_b);
    k_rnn<<<dim3(2), dim3(1024), 0, stream>>>(whhT_f, whhT_b, xw_f, xw_b, hcat);
    k_gemm_fc<<<dim3(VOCAB / 128, TLEN * BATCH / (16 * 8)), dim3(256), 0, stream>>>(hcat, wfcT, bfc, (float*)d_out);
}
